// LNGRU2dCell_90177133346850
// MI455X (gfx1250) — matrix operand order repaired and checked by reading, not compiled, not run on hardware
//
#include <hip/hip_runtime.h>

typedef __attribute__((ext_vector_type(16))) _Float16 v16h;
typedef __attribute__((ext_vector_type(8)))  _Float16 v8h;
typedef __attribute__((ext_vector_type(4)))  _Float16 v4h;
typedef __attribute__((ext_vector_type(8)))  float    v8f;

#define TM 128
#define TN 128
#define TK 32
#define LDSP 40   // padded LDS row stride in halves (32 data + 8 pad), keeps 16B alignment

// ---------------------------------------------------------------------------
// f32 -> f16 conversion (vectorized, grid-stride)
// ---------------------------------------------------------------------------
__global__ __launch_bounds__(256) void cvt_f32_to_f16(const float* __restrict__ in,
                                                      _Float16* __restrict__ out, int n4) {
  int i = blockIdx.x * blockDim.x + threadIdx.x;
  int stride = gridDim.x * blockDim.x;
  for (; i < n4; i += stride) {
    float4 v = ((const float4*)in)[i];
    v4h o;
    o[0] = (_Float16)v.x; o[1] = (_Float16)v.y;
    o[2] = (_Float16)v.z; o[3] = (_Float16)v.w;
    ((v4h*)out)[i] = o;
  }
}

// concat [s0 | s1] along last dim -> f16 (B x 2048)
__global__ __launch_bounds__(256) void concat_cvt(const float* __restrict__ s0,
                                                  const float* __restrict__ s1,
                                                  _Float16* __restrict__ out, int rows) {
  int n4 = rows * 512;  // 2048/4 float4 per row
  int i = blockIdx.x * blockDim.x + threadIdx.x;
  int stride = gridDim.x * blockDim.x;
  for (; i < n4; i += stride) {
    int row = i >> 9;
    int c4  = i & 511;
    float4 v = (c4 < 256) ? ((const float4*)s0)[row * 256 + c4]
                          : ((const float4*)s1)[row * 256 + (c4 - 256)];
    v4h o;
    o[0] = (_Float16)v.x; o[1] = (_Float16)v.y;
    o[2] = (_Float16)v.z; o[3] = (_Float16)v.w;
    ((v4h*)out)[i] = o;   // out layout matches i directly (row stride 2048 halves)
  }
}

// ---------------------------------------------------------------------------
// NT GEMM: C[M,4096] = A[M,K] * W[4096,K]^T (f16 in, f32 acc)
// v_wmma_f32_16x16x32_f16, 256 threads (8 waves), 128x128 tile, K-step 32,
// double-buffered LDS (reg-prefetch of next tile overlapped with WMMAs).
// ---------------------------------------------------------------------------
union FragU { struct { v8h lo, hi; } p; v16h v; };

template <int K>
__global__ __launch_bounds__(256) void gemm_nt_f16(const _Float16* __restrict__ A,
                                                   const _Float16* __restrict__ W,
                                                   float* __restrict__ C) {
  constexpr int N = 4096;
  __shared__ _Float16 As[2][TM * LDSP];
  __shared__ _Float16 Bs[2][TN * LDSP];

  const int t    = threadIdx.x;
  const int wave = t >> 5;
  const int lane = t & 31;
  const int wm   = wave >> 2;   // 0..1  (64-row slab)
  const int wn   = wave & 3;    // 0..3  (32-col slab)
  const int l16  = lane & 15;
  const int lhi  = lane >> 4;   // 0/1

  const int blockM = blockIdx.x * TM;
  const int blockN = blockIdx.y * TN;

  const v8f zero8 = {0.f, 0.f, 0.f, 0.f, 0.f, 0.f, 0.f, 0.f};
  v8f acc[4][2];
#pragma unroll
  for (int a = 0; a < 4; ++a)
#pragma unroll
    for (int b = 0; b < 2; ++b) acc[a][b] = zero8;

  // per-thread global->LDS loader: 512 8-half segments per 128x32 tile, 2 per thread
  const int idx0 = t, idx1 = t + 256;
  const int row0 = idx0 >> 2, seg0 = idx0 & 3;
  const int row1 = idx1 >> 2, seg1 = idx1 & 3;
  const _Float16* pa0 = A + (size_t)(blockM + row0) * K + seg0 * 8;
  const _Float16* pa1 = A + (size_t)(blockM + row1) * K + seg1 * 8;
  const _Float16* pb0 = W + (size_t)(blockN + row0) * K + seg0 * 8;
  const _Float16* pb1 = W + (size_t)(blockN + row1) * K + seg1 * 8;
  const int so0 = row0 * LDSP + seg0 * 8;
  const int so1 = row1 * LDSP + seg1 * 8;

  // per-lane fragment LDS offsets (halves), per ISA WMMA layouts
  const int aoff = l16 * LDSP + 8 * lhi;    // + wave/frag row offset added below
  const int boff = (wn * 32 + l16) * LDSP + 8 * lhi;

  // prologue: tile 0 -> LDS buffer 0
  uint4 ra0 = *(const uint4*)pa0; pa0 += TK;
  uint4 ra1 = *(const uint4*)pa1; pa1 += TK;
  uint4 rb0 = *(const uint4*)pb0; pb0 += TK;
  uint4 rb1 = *(const uint4*)pb1; pb1 += TK;
  *(uint4*)(&As[0][so0]) = ra0;
  *(uint4*)(&As[0][so1]) = ra1;
  *(uint4*)(&Bs[0][so0]) = rb0;
  *(uint4*)(&Bs[0][so1]) = rb1;
  __syncthreads();

  constexpr int NIT = K / TK;
  int buf = 0;
  for (int it = 0; it < NIT; ++it, buf ^= 1) {
    // prefetch next K-tile into registers (overlaps with WMMA below)
    if (it + 1 < NIT) {
      ra0 = *(const uint4*)pa0; pa0 += TK;
      ra1 = *(const uint4*)pa1; pa1 += TK;
      rb0 = *(const uint4*)pb0; pb0 += TK;
      rb1 = *(const uint4*)pb1; pb1 += TK;
    }

    // A fragments: lane = row l16; K in {c..c+8} U {c+16..c+24}, c = 8*lhi
    v16h afrag[4], bfrag[2];
#pragma unroll
    for (int fm = 0; fm < 4; ++fm) {
      int base = (wm * 64 + fm * 16) * LDSP + aoff;
      FragU u;
      u.p.lo = *(const v8h*)(&As[buf][base]);
      u.p.hi = *(const v8h*)(&As[buf][base + 16]);
      afrag[fm] = u.v;
    }
    // B fragments: lane = col l16; K in {c..c+8} U {c+16..c+24}, c = 8*lhi
#pragma unroll
    for (int fn = 0; fn < 2; ++fn) {
      int base = fn * (16 * LDSP) + boff;
      FragU u;
      u.p.lo = *(const v8h*)(&Bs[buf][base]);
      u.p.hi = *(const v8h*)(&Bs[buf][base + 16]);
      bfrag[fn] = u.v;
    }

#pragma unroll
    for (int fm = 0; fm < 4; ++fm)
#pragma unroll
      for (int fn = 0; fn < 2; ++fn)
        acc[fm][fn] = __builtin_amdgcn_wmma_f32_16x16x32_f16(
            false, afrag[fm], false, bfrag[fn], (short)0, acc[fm][fn], false, false);

    if (it + 1 < NIT) {
      int nb = buf ^ 1;
      *(uint4*)(&As[nb][so0]) = ra0;
      *(uint4*)(&As[nb][so1]) = ra1;
      *(uint4*)(&Bs[nb][so0]) = rb0;
      *(uint4*)(&Bs[nb][so1]) = rb1;
      __syncthreads();
    }
  }

  // writeback: C frag row = v + 8*lhi, col = l16; v*N folds into immediate offsets
#pragma unroll
  for (int fm = 0; fm < 4; ++fm) {
#pragma unroll
    for (int fn = 0; fn < 2; ++fn) {
      float* cp = C + (size_t)(blockM + wm * 64 + fm * 16 + 8 * lhi) * N
                    + (blockN + wn * 32 + fn * 16 + l16);
#pragma unroll
      for (int v = 0; v < 8; ++v) cp[(size_t)v * N] = acc[fm][fn][v];
    }
  }
}

// ---------------------------------------------------------------------------
// Per-row epilogue: LN(ig), LN(sg), gates, activations, h, LN(h)
// one 256-thread block per row b
// ---------------------------------------------------------------------------
__device__ __forceinline__ float sigm(float x) { return 1.0f / (1.0f + expf(-x)); }

__global__ __launch_bounds__(256) void epilogue_kernel(
    const float* __restrict__ ig, const float* __restrict__ sg,
    const float* __restrict__ s0, const float* __restrict__ s1,
    const float* __restrict__ lni_g, const float* __restrict__ lni_b,
    const float* __restrict__ lns_g, const float* __restrict__ lns_b,
    const float* __restrict__ lnh_g, const float* __restrict__ lnh_b,
    float* __restrict__ out) {
  __shared__ float4 sig4[1024];
  __shared__ float4 ssg4[1024];
  __shared__ float4 red4[256];
  float* sh_ig = (float*)sig4;
  float* sh_sg = (float*)ssg4;

  const int b = blockIdx.x;
  const int t = threadIdx.x;
  const float4* gi = (const float4*)(ig + (size_t)b * 4096);
  const float4* gs = (const float4*)(sg + (size_t)b * 4096);

  float si = 0.f, qi = 0.f, ss = 0.f, qs = 0.f;
#pragma unroll
  for (int p = 0; p < 4; ++p) {
    int i4 = t + p * 256;
    float4 vi = gi[i4], vs = gs[i4];
    sig4[i4] = vi; ssg4[i4] = vs;
    si += vi.x + vi.y + vi.z + vi.w;
    qi += vi.x * vi.x + vi.y * vi.y + vi.z * vi.z + vi.w * vi.w;
    ss += vs.x + vs.y + vs.z + vs.w;
    qs += vs.x * vs.x + vs.y * vs.y + vs.z * vs.z + vs.w * vs.w;
  }
  red4[t] = make_float4(si, qi, ss, qs);
  __syncthreads();
  for (int off = 128; off > 0; off >>= 1) {
    if (t < off) {
      float4 a = red4[t], c = red4[t + off];
      red4[t] = make_float4(a.x + c.x, a.y + c.y, a.z + c.z, a.w + c.w);
    }
    __syncthreads();
  }
  float4 tot = red4[0];
  const float inv = 1.0f / 4096.0f;
  float mu_i = tot.x * inv, mu_s = tot.z * inv;
  float rsi = rsqrtf(fmaxf(tot.y * inv - mu_i * mu_i, 0.f) + 1e-5f);
  float rss = rsqrtf(fmaxf(tot.w * inv - mu_s * mu_s, 0.f) + 1e-5f);

  float hval[4];
  float hs = 0.f, hq = 0.f;
#pragma unroll
  for (int p = 0; p < 4; ++p) {
    int c = t + p * 256;
#define GATE(j) (((sh_ig[j] - mu_i) * rsi) * lni_g[j] + lni_b[j] + \
                 ((sh_sg[j] - mu_s) * rss) * lns_g[j] + lns_b[j])
    float r_inv = sigm(GATE(c));
    float ii    = sigm(GATE(c + 1024));
    float npre  = GATE(c + 2048);
    float ll    = sigm(GATE(c + 3072));
#undef GATE
    float s_n = ((sh_sg[c + 2048] - mu_s) * rss) * lns_g[c + 2048] + lns_b[c + 2048];
    float nn  = tanhf(npre - r_inv * s_n);
    float p0 = s0[(size_t)b * 1024 + c];
    float p1 = s1[(size_t)b * 1024 + c];
    float h = nn + ii * (ll * p0 + (1.0f - ll) * p1 - nn);
    hval[p] = h; hs += h; hq += h * h;
  }
  __syncthreads();
  red4[t] = make_float4(hs, hq, 0.f, 0.f);
  __syncthreads();
  for (int off = 128; off > 0; off >>= 1) {
    if (t < off) {
      float4 a = red4[t], c = red4[t + off];
      red4[t] = make_float4(a.x + c.x, a.y + c.y, 0.f, 0.f);
    }
    __syncthreads();
  }
  float mu_h = red4[0].x * (1.0f / 1024.0f);
  float rsh  = rsqrtf(fmaxf(red4[0].y * (1.0f / 1024.0f) - mu_h * mu_h, 0.f) + 1e-5f);
#pragma unroll
  for (int p = 0; p < 4; ++p) {
    int c = t + p * 256;
    out[(size_t)b * 1024 + c] = (hval[p] - mu_h) * rsh * lnh_g[c] + lnh_b[c];
  }
}

// ---------------------------------------------------------------------------
extern "C" void kernel_launch(void* const* d_in, const int* in_sizes, int n_in,
                              void* d_out, int out_size, void* d_ws, size_t ws_size,
                              hipStream_t stream) {
  const float* x     = (const float*)d_in[0];
  const float* s0    = (const float*)d_in[1];
  const float* s1    = (const float*)d_in[2];
  const float* Wi    = (const float*)d_in[3];
  const float* Ws    = (const float*)d_in[4];
  const float* lni_g = (const float*)d_in[5];
  const float* lni_b = (const float*)d_in[6];
  const float* lns_g = (const float*)d_in[7];
  const float* lns_b = (const float*)d_in[8];
  const float* lnh_g = (const float*)d_in[9];
  const float* lnh_b = (const float*)d_in[10];
  float* out = (float*)d_out;

  const int D = 1024, S = 1024, G = 4096;      // 4*S
  const int B = in_sizes[0] / D;               // 16384

  // workspace layout
  _Float16* xh  = (_Float16*)d_ws;             // B*D
  _Float16* shh = xh  + (size_t)B * D;         // B*2S
  _Float16* Wih = shh + (size_t)B * 2 * S;     // G*D
  _Float16* Wsh = Wih + (size_t)G * D;         // G*2S
  float*    igb = (float*)(Wsh + (size_t)G * 2 * S);  // B*G
  float*    sgb = igb + (size_t)B * G;                // B*G

  auto nb = [](long long n) { return (unsigned)((n + 255) / 256); };

  cvt_f32_to_f16<<<nb((long long)B * D / 4), 256, 0, stream>>>(x, xh, B * D / 4);
  cvt_f32_to_f16<<<nb((long long)G * D / 4), 256, 0, stream>>>(Wi, Wih, G * D / 4);
  cvt_f32_to_f16<<<nb((long long)G * 2 * S / 4), 256, 0, stream>>>(Ws, Wsh, G * 2 * S / 4);
  concat_cvt<<<nb((long long)B * 2 * S / 4), 256, 0, stream>>>(s0, s1, shh, B);

  dim3 g1(B / TM, G / TN);
  gemm_nt_f16<1024><<<g1, 256, 0, stream>>>(xh,  Wih, igb);
  gemm_nt_f16<2048><<<g1, 256, 0, stream>>>(shh, Wsh, sgb);

  epilogue_kernel<<<B, 256, 0, stream>>>(igb, sgb, s0, s1,
                                         lni_g, lni_b, lns_g, lns_b, lnh_g, lnh_b, out);
}
